// PAB_91216515432885
// MI455X (gfx1250) — matrix operand order repaired and checked by reading, not compiled, not run on hardware
//
#include <hip/hip_runtime.h>
#include <hip/hip_bf16.h>
#include <math.h>

// ---------------------------------------------------------------------------
// PAB block for MI455X (gfx1250, wave32, WMMA).
// GEMMs: v_wmma_f32_16x16x32_bf16, double-buffered LDS tiles (1 barrier/iter),
// 2 accumulators per wave (64Mx64N block tile) sharing one B fragment.
// Global loads: unconditional float4 (global_load_b128), batched waits.
// K-tail (C=680) via clamped addresses + branchless selects.
// Memory-bound (~1.3 GB traffic, ~55 us floor @ 23.3 TB/s): sigmoid/residual
// epilogues fused into GEMMs; FFN tanh-dw-residual-product fused single pass.
// Workspace requirement: ~761 MB of d_ws.
// ---------------------------------------------------------------------------

#define DIM   256
#define HEADS 8
#define PDIM  512
#define HF    680
#define IMG   128
#define HWPX  16384   // 128*128
#define BATCH 2

typedef __bf16 bf16_t;
typedef bf16_t v16bf __attribute__((ext_vector_type(16)));
typedef bf16_t v8bf  __attribute__((ext_vector_type(8)));
typedef float  v8f   __attribute__((ext_vector_type(8)));

static __device__ __forceinline__ bf16_t f2bf(float f) { return (bf16_t)f; }

// ===========================================================================
// Pointwise-conv GEMM:  OUT[b,m,n] = sum_k W[m,k] * X[b,k,n]
// Block = 256 threads = 8 waves. Block tile 64(M) x 64(N), K-step 32.
// Each wave: two 16x16 accumulators (M and M+32) sharing one B fragment.
// Double-buffered LDS, one barrier per K-step.
// Requires N % 64 == 0 (N = 16384 here), C % 8 == 0 (256/512/680 all ok).
// epi: 0 = none, 1 = +RES, 2 = sigmoid.
// ===========================================================================
__global__ void __launch_bounds__(256)
gemm_pw_wmma(const float* __restrict__ W, const float* __restrict__ X,
             float* __restrict__ OUT, const float* __restrict__ RES,
             int O, int C, int N,
             long xBStride, long oBStride, long rBStride, int epi)
{
    __shared__ __align__(16) bf16_t lds_a[2][64][32];   // W tile, row-major   (8 KB)
    __shared__ __align__(16) bf16_t lds_bt[2][64][32];  // X tile, transposed  (8 KB)

    const int tid  = threadIdx.x;
    const int lane = tid & 31;
    const int wid  = tid >> 5;
    const int b    = blockIdx.z;
    const int mBase = blockIdx.y * 64;
    const int nBase = blockIdx.x * 64;
    const float* __restrict__ Xb = X + (long)b * xBStride;

    // A loader: 64 rows x 32 K-cols, 8 consecutive floats per thread
    const int arow   = tid >> 2;            // 0..63
    const int acol   = (tid & 3) * 8;       // 0,8,16,24
    const int mclamp = min(mBase + arow, O - 1);
    // B loader: 32 K-rows x 64 N-cols, 8 consecutive floats per thread
    const int brow = tid >> 3;              // 0..31
    const int bcol = (tid & 7) * 8;         // 0..56

    // per-wave fragment indices (ISA 7.12.2 layouts)
    const int wm  = (wid >> 2) * 16;        // 0/16; second M tile at +32
    const int wn  = (wid & 3) * 16;
    const int mfr = wm + (lane & 15);
    const int nfr = wn + (lane & 15);
    const int hiA = (lane >= 16) ? 8 : 0;
    const int hiB = (lane >= 16) ? 8 : 0;

    auto stage = [&](int k0, int buf) {
        {   // ---- A tile: W[mBase..+63, k0..+31] (all-or-none K-tail select) ----
            const int kb = k0 + acol;
            const int ks = min(kb, C - 8);          // clamped, 16B-aligned (C%8==0)
            const float* wrow = &W[(long)mclamp * C + ks];
            const float4 v0 = *(const float4*)wrow;
            const float4 v1 = *(const float4*)(wrow + 4);
            const bool ok = (kb < C);
            v8bf pa;
            pa[0] = f2bf(ok ? v0.x : 0.0f); pa[1] = f2bf(ok ? v0.y : 0.0f);
            pa[2] = f2bf(ok ? v0.z : 0.0f); pa[3] = f2bf(ok ? v0.w : 0.0f);
            pa[4] = f2bf(ok ? v1.x : 0.0f); pa[5] = f2bf(ok ? v1.y : 0.0f);
            pa[6] = f2bf(ok ? v1.z : 0.0f); pa[7] = f2bf(ok ? v1.w : 0.0f);
            *(v8bf*)&lds_a[buf][arow][acol] = pa;
        }
        {   // ---- B tile: X[k0..+31, nBase..+63] -> transposed bf16 ----
            const int k  = k0 + brow;
            const int kc = min(k, C - 1);
            const float* xrow = &Xb[(long)kc * N + nBase + bcol];
            const float4 v0 = *(const float4*)xrow;
            const float4 v1 = *(const float4*)(xrow + 4);
            const bool ok = (k < C);
            float vals[8] = {v0.x, v0.y, v0.z, v0.w, v1.x, v1.y, v1.z, v1.w};
#pragma unroll
            for (int i = 0; i < 8; ++i)
                lds_bt[buf][bcol + i][brow] = f2bf(ok ? vals[i] : 0.0f);
        }
    };

    v8f acc0 = {}, acc1 = {};
    stage(0, 0);
    __syncthreads();

    int cur = 0;
    for (int k0 = 0; k0 < C; k0 += 32) {
        const int nxt = k0 + 32;
        if (nxt < C) stage(nxt, cur ^ 1);   // prefetch next tile into other buffer

        const v8bf a0lo = *(const v8bf*)&lds_a[cur][mfr][hiA];
        const v8bf a0hi = *(const v8bf*)&lds_a[cur][mfr][16 + hiA];
        const v8bf a1lo = *(const v8bf*)&lds_a[cur][32 + mfr][hiA];
        const v8bf a1hi = *(const v8bf*)&lds_a[cur][32 + mfr][16 + hiA];
        const v8bf blo  = *(const v8bf*)&lds_bt[cur][nfr][hiB];
        const v8bf bhi  = *(const v8bf*)&lds_bt[cur][nfr][16 + hiB];
        v16bf a0, a1, bm;
#pragma unroll
        for (int i = 0; i < 8; ++i) {
            a0[i] = a0lo[i]; a0[i + 8] = a0hi[i];
            a1[i] = a1lo[i]; a1[i + 8] = a1hi[i];
            bm[i] = blo[i];  bm[i + 8] = bhi[i];
        }
        acc0 = __builtin_amdgcn_wmma_f32_16x16x32_bf16(
                   false, a0, false, bm, (short)0, acc0, false, false);
        acc1 = __builtin_amdgcn_wmma_f32_16x16x32_bf16(
                   false, a1, false, bm, (short)0, acc1, false, false);
        __syncthreads();
        cur ^= 1;
    }

    // ---- epilogue (two M tiles) ----
    const int ncol = nBase + nfr;
    const int rb   = mBase + wm + ((lane >> 4) << 3);   // D: M = r + 8*(lane>=16)
#pragma unroll
    for (int r = 0; r < 8; ++r) {
        int m0 = rb + r;
        if (m0 < O) {
            float v = acc0[r];
            if (epi == 1) v += RES[(long)b * rBStride + (long)m0 * N + ncol];
            else if (epi == 2) v = 1.0f / (1.0f + expf(-v));
            OUT[(long)b * oBStride + (long)m0 * N + ncol] = v;
        }
        int m1 = rb + 32 + r;
        if (m1 < O) {
            float v = acc1[r];
            if (epi == 1) v += RES[(long)b * rBStride + (long)m1 * N + ncol];
            else if (epi == 2) v = 1.0f / (1.0f + expf(-v));
            OUT[(long)b * oBStride + (long)m1 * N + ncol] = v;
        }
    }
}

// ===========================================================================
// Attention logits: ATT[bh, c, d] = temp[h] * sum_n q[bh,c,n] * k[bh,d,n]
// One block (4 waves, 128 thr) per (b,h). Double-buffered LDS staging with
// coalesced float4 loads (rows contiguous in n). N % 32 == 0.
// ===========================================================================
__global__ void __launch_bounds__(128)
attn_logits_wmma(const float* __restrict__ Q, const float* __restrict__ K,
                 const float* __restrict__ temp, float* __restrict__ ATT,
                 int N, long qBStride, long kBStride)
{
    __shared__ __align__(16) bf16_t lq[2][32][32];
    __shared__ __align__(16) bf16_t lk[2][32][32];

    const int tid  = threadIdx.x;
    const int lane = tid & 31;
    const int wid  = tid >> 5;
    const int bh   = blockIdx.x;
    const int b    = bh / HEADS, h = bh % HEADS;
    const float* __restrict__ q = Q + (long)b * qBStride + (long)(h * 32) * N;
    const float* __restrict__ k = K + (long)b * kBStride + (long)(h * 32) * N;

    const int lrow = tid >> 2;            // 0..31
    const int lcol = (tid & 3) * 8;       // 0,8,16,24

    const int wm  = (wid >> 1) * 16, wn = (wid & 1) * 16;
    const int mfr = wm + (lane & 15);
    const int nfr = wn + (lane & 15);
    const int hiA = (lane >= 16) ? 8 : 0;
    const int hiB = (lane >= 16) ? 8 : 0;

    auto stage = [&](int k0, int buf) {
        const float* qp = &q[(long)lrow * N + k0 + lcol];
        const float4 a0 = *(const float4*)qp;
        const float4 a1 = *(const float4*)(qp + 4);
        v8bf pq;
        pq[0] = f2bf(a0.x); pq[1] = f2bf(a0.y); pq[2] = f2bf(a0.z); pq[3] = f2bf(a0.w);
        pq[4] = f2bf(a1.x); pq[5] = f2bf(a1.y); pq[6] = f2bf(a1.z); pq[7] = f2bf(a1.w);
        *(v8bf*)&lq[buf][lrow][lcol] = pq;

        const float* kp = &k[(long)lrow * N + k0 + lcol];
        const float4 b0 = *(const float4*)kp;
        const float4 b1 = *(const float4*)(kp + 4);
        v8bf pk;
        pk[0] = f2bf(b0.x); pk[1] = f2bf(b0.y); pk[2] = f2bf(b0.z); pk[3] = f2bf(b0.w);
        pk[4] = f2bf(b1.x); pk[5] = f2bf(b1.y); pk[6] = f2bf(b1.z); pk[7] = f2bf(b1.w);
        *(v8bf*)&lk[buf][lrow][lcol] = pk;
    };

    v8f acc = {};
    stage(0, 0);
    __syncthreads();

    int cur = 0;
    for (int k0 = 0; k0 < N; k0 += 32) {
        const int nxt = k0 + 32;
        if (nxt < N) stage(nxt, cur ^ 1);

        const v8bf alo = *(const v8bf*)&lq[cur][mfr][hiA];
        const v8bf ahi = *(const v8bf*)&lq[cur][mfr][16 + hiA];
        const v8bf blo = *(const v8bf*)&lk[cur][nfr][hiB];   // B[k][d] = K[d][k]
        const v8bf bhi = *(const v8bf*)&lk[cur][nfr][16 + hiB];
        v16bf a, bm;
#pragma unroll
        for (int i = 0; i < 8; ++i) {
            a[i] = alo[i];  a[i + 8] = ahi[i];
            bm[i] = blo[i]; bm[i + 8] = bhi[i];
        }
        acc = __builtin_amdgcn_wmma_f32_16x16x32_bf16(
                  false, a, false, bm, (short)0, acc, false, false);
        __syncthreads();
        cur ^= 1;
    }

    const float t = temp[h];
    const int rbase = wm + ((lane >> 4) << 3);
#pragma unroll
    for (int r = 0; r < 8; ++r) {
        int m = rbase + r;
        ATT[((long)bh * 32 + m) * 32 + nfr] = acc[r] * t;
    }
}

// Row softmax over 32 elements: one wave per row, 8 rows per block.
__global__ void __launch_bounds__(256)
softmax32(float* __restrict__ ATT, int rows)
{
    const int lane = threadIdx.x & 31;
    const int row  = blockIdx.x * 8 + (threadIdx.x >> 5);
    if (row >= rows) return;
    float v = ATT[row * 32 + lane];
    float m = v;
    for (int o = 16; o > 0; o >>= 1) m = fmaxf(m, __shfl_xor(m, o, 32));
    float e = expf(v - m);
    float s = e;
    for (int o = 16; o > 0; o >>= 1) s += __shfl_xor(s, o, 32);
    ATT[row * 32 + lane] = e / s;
}

// ===========================================================================
// Attention apply: OUT[b, h*32+c, n] = sum_d ATT[bh,c,d] * V[bh,d,n]
// K = 32 -> single WMMA step. Block tile 32(M) x 64(N) per (b,h).
// ===========================================================================
__global__ void __launch_bounds__(256)
attn_apply_wmma(const float* __restrict__ ATT, const float* __restrict__ V,
                float* __restrict__ OUT, int N, long vBStride, long oBStride)
{
    const int lane = threadIdx.x & 31;
    const int wid  = threadIdx.x >> 5;
    const int bh   = blockIdx.y;
    const int b    = bh / HEADS, h = bh % HEADS;
    const float* __restrict__ att = ATT + (long)bh * 32 * 32;
    const float* __restrict__ v   = V + (long)b * vBStride + (long)(h * 32) * N;

    const int mTile = (wid >> 2) * 16;
    const int nBase = blockIdx.x * 64 + (wid & 3) * 16;
    const int mr  = mTile + (lane & 15);
    const int nc  = nBase + (lane & 15);
    const int hiA = (lane >= 16) ? 8 : 0;
    const int hiB = (lane >= 16) ? 8 : 0;

    // A fragment: two 8-float runs (16B-aligned), L0-resident 4 KB table
    const float4 a0 = *(const float4*)&att[mr * 32 + hiA];
    const float4 a1 = *(const float4*)&att[mr * 32 + hiA + 4];
    const float4 a2 = *(const float4*)&att[mr * 32 + 16 + hiA];
    const float4 a3 = *(const float4*)&att[mr * 32 + 16 + hiA + 4];
    v16bf a;
    a[0]  = f2bf(a0.x); a[1]  = f2bf(a0.y); a[2]  = f2bf(a0.z); a[3]  = f2bf(a0.w);
    a[4]  = f2bf(a1.x); a[5]  = f2bf(a1.y); a[6]  = f2bf(a1.z); a[7]  = f2bf(a1.w);
    a[8]  = f2bf(a2.x); a[9]  = f2bf(a2.y); a[10] = f2bf(a2.z); a[11] = f2bf(a2.w);
    a[12] = f2bf(a3.x); a[13] = f2bf(a3.y); a[14] = f2bf(a3.z); a[15] = f2bf(a3.w);

    v16bf bm;
#pragma unroll
    for (int hh = 0; hh < 16; ++hh) {
        int kb = hiB + (hh & 7) + ((hh >> 3) << 4);
        bm[hh] = f2bf(v[(long)kb * N + nc]);
    }
    v8f acc = {};
    acc = __builtin_amdgcn_wmma_f32_16x16x32_bf16(
              false, a, false, bm, (short)0, acc, false, false);

    const int rbase = mTile + ((lane >> 4) << 3);
#pragma unroll
    for (int r = 0; r < 8; ++r) {
        int m = rbase + r;
        OUT[(long)b * oBStride + (long)(h * 32 + m) * N + nc] = acc[r];
    }
}

// ===========================================================================
// Elementwise / depthwise / normalization kernels
// ===========================================================================
__global__ void __launch_bounds__(256)
dwconv3x3(const float* __restrict__ in, const float* __restrict__ w,
          float* __restrict__ out, int C, int H, int Wd, long total)
{
    long idx = (long)blockIdx.x * 256 + threadIdx.x;
    if (idx >= total) return;
    int x = (int)(idx % Wd);
    long t = idx / Wd;
    int y = (int)(t % H); t /= H;
    int c = (int)(t % C);
    long b = t / C;
    const float* ip = in + ((b * C + c) * (long)H) * Wd;
    const float* wp = w + (long)c * 9;
    float s = 0.f;
#pragma unroll
    for (int ky = 0; ky < 3; ++ky) {
        int yy = y + ky - 1;
        if (yy < 0 || yy >= H) continue;
#pragma unroll
        for (int kx = 0; kx < 3; ++kx) {
            int xx = x + kx - 1;
            if (xx < 0 || xx >= Wd) continue;
            s += wp[ky * 3 + kx] * ip[(long)yy * Wd + xx];
        }
    }
    out[idx] = s;
}

// Fused FFN tail: x1=tanh(dw1(h[0:HF]))+h[0:HF]; x2=tanh(dw2(h[HF:]))+h[HF:];
// prod = x1*x2 — single pass over the 178 MB hidden tensor.
__global__ void __launch_bounds__(256)
ffn_tanh_prod(const float* __restrict__ hbuf, const float* __restrict__ w1,
              const float* __restrict__ w2, float* __restrict__ prod, long total)
{
    long idx = (long)blockIdx.x * 256 + threadIdx.x;   // over B*HF*HWPX
    if (idx >= total) return;
    int x = (int)(idx % IMG);
    long t = idx / IMG;
    int y = (int)(t % IMG); t /= IMG;
    int c = (int)(t % HF);
    long b = t / HF;
    const float* p1 = hbuf + (b * (2 * HF) + c) * (long)HWPX;
    const float* p2 = hbuf + (b * (2 * HF) + HF + c) * (long)HWPX;
    const float* w1p = w1 + (long)c * 9;
    const float* w2p = w2 + (long)c * 9;
    float s1 = 0.f, s2 = 0.f;
#pragma unroll
    for (int ky = 0; ky < 3; ++ky) {
        int yy = y + ky - 1;
        if (yy < 0 || yy >= IMG) continue;
#pragma unroll
        for (int kx = 0; kx < 3; ++kx) {
            int xx = x + kx - 1;
            if (xx < 0 || xx >= IMG) continue;
            s1 += w1p[ky * 3 + kx] * p1[(long)yy * IMG + xx];
            s2 += w2p[ky * 3 + kx] * p2[(long)yy * IMG + xx];
        }
    }
    float c1 = tanhf(s1) + p1[(long)y * IMG + x];
    float c2 = tanhf(s2) + p2[(long)y * IMG + x];
    prod[(b * HF + c) * (long)HWPX + (long)y * IMG + x] = c1 * c2;
}

// Channel LayerNorm (Restormer WithBias): one thread per (b,pixel).
__global__ void __launch_bounds__(256)
layernorm_c(const float* __restrict__ in, const float* __restrict__ g,
            const float* __restrict__ bta, float* __restrict__ out,
            int C, long total)
{
    long idx = (long)blockIdx.x * 256 + threadIdx.x;   // b*HWPX + pix
    if (idx >= total) return;
    long b = idx / HWPX;
    long pix = idx % HWPX;
    const float* ip = in + b * (long)C * HWPX + pix;
    float s = 0.f, ss = 0.f;
    for (int c = 0; c < C; ++c) { float v = ip[(long)c * HWPX]; s += v; ss += v * v; }
    float mu = s / (float)C;
    float var = ss / (float)C - mu * mu;
    float rstd = rsqrtf(var + 1e-5f);
    float* op = out + b * (long)C * HWPX + pix;
    for (int c = 0; c < C; ++c) {
        float v = (ip[(long)c * HWPX] - mu) * rstd;
        op[(long)c * HWPX] = v * g[c] + bta[c];
    }
}

__global__ void __launch_bounds__(256)
modulate(const float* __restrict__ x, const float* __restrict__ scale,
         const float* __restrict__ shift, float* __restrict__ out, long total)
{
    long idx = (long)blockIdx.x * 256 + threadIdx.x;
    if (idx >= total) return;
    long t = idx / HWPX;
    int c = (int)(t % DIM);
    long b = t / DIM;
    float sc = scale[b * DIM + c];
    float sh = shift[b * DIM + c];
    out[idx] = x[idx] * (1.0f + sc) + sh;
}

__global__ void __launch_bounds__(256)
gate_mix(const float* __restrict__ q12, const float* __restrict__ gate,
         float* __restrict__ q, long total)
{
    long idx = (long)blockIdx.x * 256 + threadIdx.x;   // [b,256,HWPX]
    if (idx >= total) return;
    long pix = idx % HWPX;
    long t = idx / HWPX;
    int c = (int)(t % DIM);
    long b = t / DIM;
    float g  = gate[idx];
    float q1 = q12[(b * 2 * DIM + c) * (long)HWPX + pix];
    float q2 = q12[(b * 2 * DIM + DIM + c) * (long)HWPX + pix];
    q[idx] = g * q1 + (1.0f - g) * q2;
}

// L2-normalize rows of length N (F.normalize(dim=-1), eps=1e-12), in place.
__global__ void __launch_bounds__(256)
l2norm_rows(float* __restrict__ buf, int rowsPerBatch, long bstride, int N)
{
    __shared__ float red[256];
    int row = blockIdx.x;
    int b = row / rowsPerBatch, r = row % rowsPerBatch;
    float* p = buf + (long)b * bstride + (long)r * N;
    float s = 0.f;
    for (int i = threadIdx.x; i < N; i += 256) { float v = p[i]; s += v * v; }
    red[threadIdx.x] = s;
    __syncthreads();
    for (int o = 128; o > 0; o >>= 1) {
        if ((int)threadIdx.x < o) red[threadIdx.x] += red[threadIdx.x + o];
        __syncthreads();
    }
    float inv = 1.0f / fmaxf(sqrtf(red[0]), 1e-12f);
    for (int i = threadIdx.x; i < N; i += 256) p[i] *= inv;
}

// scale = sigmoid(prompt @ scale_w^T + scale_b); shift = prompt @ shift_w^T + shift_b
__global__ void __launch_bounds__(256)
prompt_vecs(const float* __restrict__ prompt,
            const float* __restrict__ sw, const float* __restrict__ sb,
            const float* __restrict__ hw_, const float* __restrict__ hb,
            float* __restrict__ scale, float* __restrict__ shift)
{
    int i = blockIdx.x * 256 + threadIdx.x;
    if (i >= BATCH * DIM) return;
    int b = i / DIM, o = i % DIM;
    const float* p = prompt + (long)b * PDIM;
    float s1 = sb[o], s2 = hb[o];
    for (int k = 0; k < PDIM; ++k) {
        float pv = p[k];
        s1 += pv * sw[(long)o * PDIM + k];
        s2 += pv * hw_[(long)o * PDIM + k];
    }
    scale[i] = 1.0f / (1.0f + expf(-s1));
    shift[i] = s2;
}

__global__ void __launch_bounds__(256)
ps_linear(const float* __restrict__ prompt, const float* __restrict__ w,
          const float* __restrict__ bia, float* __restrict__ out, long total)
{
    long i = (long)blockIdx.x * 256 + threadIdx.x;   // B * DIM*16*16
    if (i >= total) return;
    long b = i / (DIM * 256);
    long o = i % (DIM * 256);
    const float* p = prompt + b * PDIM;
    float s = bia[o];
    for (int k = 0; k < PDIM; ++k) s += p[k] * w[o * PDIM + k];
    out[i] = s;
}

// Full 3x3 conv 256->256 on 16x16 + exact GELU.
__global__ void __launch_bounds__(256)
ps_conv_gelu(const float* __restrict__ in, const float* __restrict__ w,
             float* __restrict__ out, long total)
{
    long idx = (long)blockIdx.x * 256 + threadIdx.x;   // B*DIM*16*16
    if (idx >= total) return;
    int x = (int)(idx % 16);
    long t = idx / 16;
    int y = (int)(t % 16); t /= 16;
    int o = (int)(t % DIM);
    long b = t / DIM;
    const float* ib = in + b * (long)DIM * 256;
    float s = 0.f;
    for (int c = 0; c < DIM; ++c) {
        const float* ip = ib + (long)c * 256;
        const float* wp = w + ((long)o * DIM + c) * 9;
#pragma unroll
        for (int ky = 0; ky < 3; ++ky) {
            int yy = y + ky - 1;
            if (yy < 0 || yy >= 16) continue;
#pragma unroll
            for (int kx = 0; kx < 3; ++kx) {
                int xx = x + kx - 1;
                if (xx < 0 || xx >= 16) continue;
                s += wp[ky * 3 + kx] * ip[yy * 16 + xx];
            }
        }
    }
    out[idx] = 0.5f * s * (1.0f + erff(s * 0.70710678118654752f));
}

// Bilinear 16 -> 128 (align_corners=False, half-pixel centers, edge clamp).
__global__ void __launch_bounds__(256)
upsample_bilinear(const float* __restrict__ in, float* __restrict__ out, long total)
{
    long idx = (long)blockIdx.x * 256 + threadIdx.x;   // B*DIM*128*128
    if (idx >= total) return;
    int X = (int)(idx % IMG);
    long t = idx / IMG;
    int Y = (int)(t % IMG); t /= IMG;
    int c = (int)(t % DIM);
    long b = t / DIM;
    float sy = (Y + 0.5f) * 0.125f - 0.5f;
    float sx = (X + 0.5f) * 0.125f - 0.5f;
    int y0 = (int)floorf(sy), x0 = (int)floorf(sx);
    float wy = sy - y0, wx = sx - x0;
    int y0c = min(max(y0, 0), 15), y1c = min(max(y0 + 1, 0), 15);
    int x0c = min(max(x0, 0), 15), x1c = min(max(x0 + 1, 0), 15);
    const float* ip = in + (b * DIM + c) * 256;
    float v00 = ip[y0c * 16 + x0c], v01 = ip[y0c * 16 + x1c];
    float v10 = ip[y1c * 16 + x0c], v11 = ip[y1c * 16 + x1c];
    float top = v00 + (v01 - v00) * wx;
    float bot = v10 + (v11 - v10) * wx;
    out[idx] = top + (bot - top) * wy;
}

// ===========================================================================
// Host orchestration
// ===========================================================================
extern "C" void kernel_launch(void* const* d_in, const int* in_sizes, int n_in,
                              void* d_out, int out_size, void* d_ws, size_t ws_size,
                              hipStream_t stream)
{
    (void)in_sizes; (void)n_in; (void)out_size; (void)ws_size;

    const float* x         = (const float*)d_in[0];
    const float* y         = (const float*)d_in[1];
    const float* prompt    = (const float*)d_in[2];
    const float* ln_w      = (const float*)d_in[3];
    const float* ln_b      = (const float*)d_in[4];
    const float* scale_w   = (const float*)d_in[5];
    const float* scale_b   = (const float*)d_in[6];
    const float* shift_w   = (const float*)d_in[7];
    const float* shift_b   = (const float*)d_in[8];
    const float* ps_lin_w  = (const float*)d_in[9];
    const float* ps_lin_b  = (const float*)d_in[10];
    const float* ps_conv_w = (const float*)d_in[11];
    const float* ps_dw_w   = (const float*)d_in[12];
    const float* temperature = (const float*)d_in[13];
    const float* q1_w      = (const float*)d_in[14];
    const float* q2_w      = (const float*)d_in[15];
    const float* q_dw_w    = (const float*)d_in[16];
    const float* gate_dw_w = (const float*)d_in[17];
    const float* gate_pw_w = (const float*)d_in[18];
    const float* kv_w      = (const float*)d_in[19];
    const float* kv_dw_w   = (const float*)d_in[20];
    const float* proj_w    = (const float*)d_in[21];
    const float* ffn_in_w  = (const float*)d_in[22];
    const float* ffn_dw_w  = (const float*)d_in[23];
    const float* ffn_dw1_w = (const float*)d_in[24];
    const float* ffn_dw2_w = (const float*)d_in[25];
    const float* ffn_out_w = (const float*)d_in[26];

    float* ws = (float*)d_ws;
    const int  N  = HWPX;                                // 16384
    const long S1 = (long)BATCH * DIM * N;               // 8.4M floats (33.5 MB)
    const long S2 = 2 * S1;                              // 67 MB
    const long SH = (long)BATCH * 2 * HF * N;            // 178 MB

    // small scratch
    float* scale  = ws;
    float* shiftv = ws + 512;
    float* pslin  = ws + 1024;            // 131072
    float* pfa    = pslin + 131072;
    float* pfb    = pfa + 131072;
    float* attn   = pfb + 131072;         // 16384
    // big scratch (sequential reuse noted per buffer)
    float* big  = ws + (1 << 19);
    float* pf   = big;          // prompt feature upsampled
    float* f1   = pf  + S1;     // x_mod  -> later x_after_attn
    float* pn   = f1  + S1;     // LN(pf)
    float* xn   = pn  + S1;     // LN(x_mod) -> later LN(x_after)
    float* yn   = xn  + S1;     // LN(y)
    float* gate = yn  + S1;     // sigmoid gate -> later attn_out
    float* qmix = gate + S1;    // mixed q
    float* qd   = qmix + S1;    // dw(q), l2-normalized in place
    float* q12  = qd  + S1;     // [q1;q2] (S2) -> later kv_dw output (k,v)
    float* g512 = q12 + S2;     // gate_dw out (S2) -> later kv_pw output
    float* h0   = g512 + S2;    // FFN hidden pre-dw (SH) -> later prod
    float* h1   = h0  + SH;     // FFN hidden post-dw (SH)

    const long lnTot = (long)BATCH * N;
    const dim3 blk(256);
    const int gN = N / 64;                               // 256

    // ---- prompt modulation + prompt spatial feature ----
    prompt_vecs<<<2, blk, 0, stream>>>(prompt, scale_w, scale_b, shift_w, shift_b,
                                       scale, shiftv);
    modulate<<<(unsigned)((S1 + 255) / 256), blk, 0, stream>>>(x, scale, shiftv, f1, S1);
    ps_linear<<<(131072 + 255) / 256, blk, 0, stream>>>(prompt, ps_lin_w, ps_lin_b,
                                                        pslin, 131072);
    ps_conv_gelu<<<(131072 + 255) / 256, blk, 0, stream>>>(pslin, ps_conv_w, pfa, 131072);
    dwconv3x3<<<(131072 + 255) / 256, blk, 0, stream>>>(pfa, ps_dw_w, pfb, DIM, 16, 16, 131072);
    upsample_bilinear<<<(unsigned)((S1 + 255) / 256), blk, 0, stream>>>(pfb, pf, S1);

    // ---- LayerNorms ----
    layernorm_c<<<(unsigned)((lnTot + 255) / 256), blk, 0, stream>>>(pf, ln_w, ln_b, pn, DIM, lnTot);
    layernorm_c<<<(unsigned)((lnTot + 255) / 256), blk, 0, stream>>>(f1, ln_w, ln_b, xn, DIM, lnTot);
    layernorm_c<<<(unsigned)((lnTot + 255) / 256), blk, 0, stream>>>(y,  ln_w, ln_b, yn, DIM, lnTot);

    // ---- q path: q1 = pw(pn), q2 = pw(xn) into concat buffer ----
    gemm_pw_wmma<<<dim3(gN, (DIM + 63) / 64, BATCH), blk, 0, stream>>>(
        q1_w, pn, q12, nullptr, DIM, DIM, N, (long)DIM * N, (long)2 * DIM * N, 0, 0);
    gemm_pw_wmma<<<dim3(gN, (DIM + 63) / 64, BATCH), blk, 0, stream>>>(
        q2_w, xn, q12 + (long)DIM * N, nullptr, DIM, DIM, N, (long)DIM * N, (long)2 * DIM * N, 0, 0);
    dwconv3x3<<<(unsigned)((S2 + 255) / 256), blk, 0, stream>>>(
        q12, gate_dw_w, g512, 2 * DIM, IMG, IMG, S2);
    gemm_pw_wmma<<<dim3(gN, (DIM + 63) / 64, BATCH), blk, 0, stream>>>(
        gate_pw_w, g512, gate, nullptr, DIM, 2 * DIM, N, (long)2 * DIM * N, (long)DIM * N, 0, 2);
    gate_mix<<<(unsigned)((S1 + 255) / 256), blk, 0, stream>>>(q12, gate, qmix, S1);
    dwconv3x3<<<(unsigned)((S1 + 255) / 256), blk, 0, stream>>>(
        qmix, q_dw_w, qd, DIM, IMG, IMG, S1);

    // ---- kv path: kv = dw(pw(yn)); k = rows 0..255, v = rows 256..511 ----
    gemm_pw_wmma<<<dim3(gN, (2 * DIM + 63) / 64, BATCH), blk, 0, stream>>>(
        kv_w, yn, g512, nullptr, 2 * DIM, DIM, N, (long)DIM * N, (long)2 * DIM * N, 0, 0);
    dwconv3x3<<<(unsigned)((S2 + 255) / 256), blk, 0, stream>>>(
        g512, kv_dw_w, q12, 2 * DIM, IMG, IMG, S2);          // kvd lives in q12 now

    // ---- channel attention ----
    l2norm_rows<<<BATCH * DIM, blk, 0, stream>>>(qd,  DIM, (long)DIM * N, N);
    l2norm_rows<<<BATCH * DIM, blk, 0, stream>>>(q12, DIM, (long)2 * DIM * N, N);  // k rows only
    attn_logits_wmma<<<BATCH * HEADS, 128, 0, stream>>>(
        qd, q12, temperature, attn, N, (long)DIM * N, (long)2 * DIM * N);
    softmax32<<<(BATCH * HEADS * 32) / 8, blk, 0, stream>>>(attn, BATCH * HEADS * 32);
    attn_apply_wmma<<<dim3(gN, BATCH * HEADS), blk, 0, stream>>>(
        attn, q12 + (long)DIM * N, gate, N, (long)2 * DIM * N, (long)DIM * N);  // out -> gate buf
    // x_after = x + pw(attn_out)   (residual fused in epilogue)
    gemm_pw_wmma<<<dim3(gN, (DIM + 63) / 64, BATCH), blk, 0, stream>>>(
        proj_w, gate, f1, x, DIM, DIM, N, (long)DIM * N, (long)DIM * N, (long)DIM * N, 1);

    // ---- FFN ----
    layernorm_c<<<(unsigned)((lnTot + 255) / 256), blk, 0, stream>>>(f1, ln_w, ln_b, xn, DIM, lnTot);
    gemm_pw_wmma<<<dim3(gN, (2 * HF + 63) / 64, BATCH), blk, 0, stream>>>(
        ffn_in_w, xn, h0, nullptr, 2 * HF, DIM, N, (long)DIM * N, (long)2 * HF * N, 0, 0);
    dwconv3x3<<<(unsigned)((SH + 255) / 256), blk, 0, stream>>>(
        h0, ffn_dw_w, h1, 2 * HF, IMG, IMG, SH);
    ffn_tanh_prod<<<(unsigned)((SH / 2 + 255) / 256), blk, 0, stream>>>(
        h1, ffn_dw1_w, ffn_dw2_w, h0, SH / 2);               // prod reuses h0
    // out = x_after + pw(prod)
    gemm_pw_wmma<<<dim3(gN, (DIM + 63) / 64, BATCH), blk, 0, stream>>>(
        ffn_out_w, h0, (float*)d_out, f1, DIM, HF, N, (long)HF * N, (long)DIM * N, (long)DIM * N, 1);
}
